// FullAttentionBlock_45724221833278
// MI455X (gfx1250) — matrix operand order repaired and checked by reading, not compiled, not run on hardware
//
#include <hip/hip_runtime.h>
#include <hip/hip_bf16.h>

// ---------------------------------------------------------------------------
// FullAttentionBlock for MI455X (gfx1250): bf16 WMMA GEMMs + WMMA flash attn
// Async global->LDS staging (ASYNCcnt), double-buffered GEMM + attention.
// B=4, N=1024, D=1024, H=16, HD=64, HID=4096
// ---------------------------------------------------------------------------

typedef __bf16 bf16;
typedef __bf16 v16bf __attribute__((ext_vector_type(16)));
typedef float  v8f   __attribute__((ext_vector_type(8)));
typedef unsigned int u32x4 __attribute__((ext_vector_type(4)));

union FragU { u32x4 u[2]; v16bf v; };

#define BB   4
#define NN   1024
#define DD   1024
#define HH   16
#define HDD  64
#define HIDD 4096
#define BNROWS 4096   // B*N

__device__ __forceinline__ v8f wmma_bf16(v16bf a, v16bf b, v8f c) {
  // D = A(16x32) * B(32x16) + C, f32 accumulate
  return __builtin_amdgcn_wmma_f32_16x16x32_bf16(false, a, false, b, (short)0, c,
                                                 false, false);
}

// 16B global -> LDS async copy (ASYNCcnt tracked). LDS offset = low 32 bits of
// the generic pointer (flat LDS aperture keeps addr[31:0] = LDS byte offset).
__device__ __forceinline__ void async_b128(void* lds_dst, const void* gsrc) {
  unsigned lds = (unsigned)(unsigned long long)lds_dst;
  asm volatile("global_load_async_to_lds_b128 %0, %1, off"
               :: "v"(lds), "v"(gsrc) : "memory");
}
__device__ __forceinline__ void wait_async0() {
  asm volatile("s_wait_asynccnt 0x0" ::: "memory");
}

// ---------------------------------------------------------------------------
// fp32 -> bf16 conversion (weights, once per launch)
// ---------------------------------------------------------------------------
__global__ void f2bf_kernel(const float* __restrict__ in, bf16* __restrict__ out, int n) {
  int i = blockIdx.x * blockDim.x + threadIdx.x;
  int stride = gridDim.x * blockDim.x;
  for (; i < n; i += stride) out[i] = (bf16)in[i];
}

// ---------------------------------------------------------------------------
// LayerNorm over D=1024, one block per row, bf16 output
// ---------------------------------------------------------------------------
__global__ __launch_bounds__(256) void ln_kernel(const float* __restrict__ x,
                                                 const float* __restrict__ w,
                                                 const float* __restrict__ b,
                                                 bf16* __restrict__ out) {
  __shared__ float red[256];
  __shared__ float red2[256];
  const int row = blockIdx.x;
  const float* xr = x + (size_t)row * DD;
  const int tid = threadIdx.x;
  float s = 0.f, s2 = 0.f;
  for (int i = tid; i < DD; i += 256) { float v = xr[i]; s += v; s2 += v * v; }
  red[tid] = s; red2[tid] = s2;
  __syncthreads();
  for (int off = 128; off > 0; off >>= 1) {
    if (tid < off) { red[tid] += red[tid + off]; red2[tid] += red2[tid + off]; }
    __syncthreads();
  }
  const float mean = red[0] * (1.f / DD);
  const float var  = red2[0] * (1.f / DD) - mean * mean;
  const float inv  = rsqrtf(var + 1e-6f);
  for (int i = tid; i < DD; i += 256)
    out[(size_t)row * DD + i] = (bf16)((xr[i] - mean) * inv * w[i] + b[i]);
}

// ---------------------------------------------------------------------------
// Tiled bf16 GEMM: C[M,Nd] = A[M,K] @ W[Nd,K]^T + bias
//   MODE 0: f32 out   MODE 1: bf16 out   MODE 2: f32 out = res + gamma*(acc+bias)
// Block 256 thr (8 waves), tile 128x128, K-step 32; wave = 64x32 (4x2 wmma tiles)
// Double-buffered LDS fed by global_load_async_to_lds_b128.
// ---------------------------------------------------------------------------
template <int MODE>
__global__ __launch_bounds__(256) void gemm_bf16_kernel(
    const bf16* __restrict__ A, const bf16* __restrict__ W,
    const float* __restrict__ bias, void* __restrict__ Cout,
    const float* __restrict__ res, const float* __restrict__ gamma,
    int M, int Nd, int K) {
  __shared__ __align__(16) bf16 a_s[2][128 * 32];
  __shared__ __align__(16) bf16 w_s[2][128 * 32];
  const int m0 = blockIdx.y * 128;
  const int n0 = blockIdx.x * 128;
  const int tid = threadIdx.x;
  const int wave = tid >> 5, lane = tid & 31;
  const int wm = wave & 1, wn = wave >> 1;     // 2 waves along M, 4 along N
  const int lh = lane & 15, ls = lane >> 4;

  v8f acc[4][2];
  for (int mt = 0; mt < 4; mt++)
    for (int nt = 0; nt < 2; nt++)
      for (int r = 0; r < 8; r++) acc[mt][nt][r] = 0.f;

  const int ldr = tid >> 1;            // 0..127
  const int ldc = (tid & 1) * 16;      // 0 or 16 (halves)

  auto stage = [&](int kt, int bufi) {
    const bf16* srcA = A + (size_t)(m0 + ldr) * K + kt * 32 + ldc;
    const bf16* srcW = W + (size_t)(n0 + ldr) * K + kt * 32 + ldc;
    bf16* da = a_s[bufi] + ldr * 32 + ldc;
    bf16* dw = w_s[bufi] + ldr * 32 + ldc;
    async_b128(da,     srcA);
    async_b128(da + 8, srcA + 8);
    async_b128(dw,     srcW);
    async_b128(dw + 8, srcW + 8);
  };

  const int nk = K >> 5;
  stage(0, 0);
  for (int kt = 0; kt < nk; ++kt) {
    const int bufi = kt & 1;
    wait_async0();        // my tile-kt copies (and older) landed in LDS
    __syncthreads();      // everyone's copies landed; prior reads of buf^1 done
    if (kt + 1 < nk) stage(kt + 1, bufi ^ 1);

    v16bf af[4], wf[2];
    for (int mt = 0; mt < 4; mt++) {
      const bf16* p = a_s[bufi] + (wm * 64 + mt * 16 + lh) * 32 + ls * 8;
      FragU f; f.u[0] = *(const u32x4*)p; f.u[1] = *(const u32x4*)(p + 16);
      af[mt] = f.v;
    }
    for (int nt = 0; nt < 2; nt++) {
      const bf16* p = w_s[bufi] + (wn * 32 + nt * 16 + lh) * 32 + ls * 8;
      FragU f; f.u[0] = *(const u32x4*)p; f.u[1] = *(const u32x4*)(p + 16);
      wf[nt] = f.v;
    }
    for (int mt = 0; mt < 4; mt++)
      for (int nt = 0; nt < 2; nt++)
        acc[mt][nt] = wmma_bf16(af[mt], wf[nt], acc[mt][nt]);
  }

  for (int mt = 0; mt < 4; mt++) {
    for (int nt = 0; nt < 2; nt++) {
      const int nb = n0 + wn * 32 + nt * 16 + lh;
      const float bn = bias[nb];
      for (int r = 0; r < 8; r++) {
        const int m = m0 + wm * 64 + mt * 16 + ls * 8 + r;
        const float val = acc[mt][nt][r] + bn;
        const size_t idx = (size_t)m * Nd + nb;
        if (MODE == 0)      ((float*)Cout)[idx] = val;
        else if (MODE == 1) ((bf16*)Cout)[idx] = (bf16)val;
        else                ((float*)Cout)[idx] = res[idx] + gamma[nb] * val;
      }
    }
  }
}

// ---------------------------------------------------------------------------
// QKV post-process: per-head RMSNorm + RoPE + spatial projections.
// Writes q'(128) = [scale*rope(rms(q)); scale*sq], k'(128) = [rope(rms(k)); sk]
// in (b,h,n,*) layout, and V TRANSPOSED as (b,h,dv,n) so the attention kernel
// can stage it with plain contiguous async copies. One wave per (b,h,n) row.
// ---------------------------------------------------------------------------
__global__ __launch_bounds__(256) void qkv_post_kernel(
    const float* __restrict__ qkv, const int* __restrict__ rope_pos,
    const float* __restrict__ qnw, const float* __restrict__ knw,
    const float* __restrict__ coords, const float* __restrict__ sqw,
    const float* __restrict__ skw, bf16* __restrict__ qp, bf16* __restrict__ kp,
    bf16* __restrict__ vtout) {
  const int tid = threadIdx.x;
  const int lane = tid & 31;
  const int rowid = blockIdx.x * 8 + (tid >> 5);  // 0..65535 = bh*N + n
  const int n = rowid & (NN - 1);
  const int bh = rowid >> 10;
  const int b = bh >> 4, h = bh & 15;

  const float* qrow = qkv + (size_t)(b * NN + n) * (3 * DD) + h * HDD;
  const float* krow = qrow + DD;
  const float* vrow = qrow + 2 * DD;
  const int i0 = lane * 2;

  float q1 = qrow[i0], q2 = qrow[i0 + 1];
  float k1 = krow[i0], k2 = krow[i0 + 1];
  float ssq = q1 * q1 + q2 * q2;
  float ssk = k1 * k1 + k2 * k2;
  for (int off = 16; off > 0; off >>= 1) {
    ssq += __shfl_xor(ssq, off, 32);
    ssk += __shfl_xor(ssk, off, 32);
  }
  const float qs = rsqrtf(ssq * (1.f / HDD) + 1e-6f);
  const float ks = rsqrtf(ssk * (1.f / HDD) + 1e-6f);
  q1 *= qs * qnw[i0]; q2 *= qs * qnw[i0 + 1];
  k1 *= ks * knw[i0]; k2 *= ks * knw[i0 + 1];

  const float pos = (float)rope_pos[n];
  const float fr = __powf(10000.f, -(float)lane * (1.f / 32.f));
  const float ang = pos * fr;
  const float sn = __sinf(ang), cs = __cosf(ang);
  const float yq1 = q1 * cs - q2 * sn, yq2 = q1 * sn + q2 * cs;
  const float yk1 = k1 * cs - k2 * sn, yk2 = k1 * sn + k2 * cs;

  const float scale = 0.125f;  // 1/sqrt(HD)
  const size_t qb = ((size_t)bh * NN + n) * 128;
  qp[qb + i0]     = (bf16)(yq1 * scale);
  qp[qb + i0 + 1] = (bf16)(yq2 * scale);
  kp[qb + i0]     = (bf16)yk1;
  kp[qb + i0 + 1] = (bf16)yk2;

  const float* cd = coords + (size_t)(b * NN + n) * 3;
  const float c0 = cd[0], c1 = cd[1], c2 = cd[2];
  const int d0 = h * HDD + i0;
  const float* w0 = sqw + (size_t)d0 * 3;
  const float sqa = c0 * w0[0] + c1 * w0[1] + c2 * w0[2];
  const float sqb = c0 * w0[3] + c1 * w0[4] + c2 * w0[5];
  const float* w1 = skw + (size_t)d0 * 3;
  const float ska = c0 * w1[0] + c1 * w1[1] + c2 * w1[2];
  const float skb = c0 * w1[3] + c1 * w1[4] + c2 * w1[5];
  qp[qb + 64 + i0]     = (bf16)(sqa * scale);   // SP_SCALE = 1
  qp[qb + 64 + i0 + 1] = (bf16)(sqb * scale);
  kp[qb + 64 + i0]     = (bf16)ska;
  kp[qb + 64 + i0 + 1] = (bf16)skb;

  // V transposed: vt[(bh*64 + dv)*N + n]
  const size_t vb = (size_t)bh * 64 * NN;
  vtout[vb + (size_t)i0 * NN + n]       = (bf16)vrow[i0];
  vtout[vb + (size_t)(i0 + 1) * NN + n] = (bf16)vrow[i0 + 1];
}

// ---------------------------------------------------------------------------
// Flash attention (non-causal): one WG per (b,h, 128 q-rows), 8 waves x 16 rows.
// Feature dim 128 (q/k + spatial bias folded in), dv = 64, key block 64.
// K and V^T blocks double-buffered via async-to-LDS: block kb+1 streams in
// while block kb is consumed; one barrier per block.
// ---------------------------------------------------------------------------
__global__ __launch_bounds__(256) void attn_kernel(const bf16* __restrict__ qp,
                                                   const bf16* __restrict__ kpg,
                                                   const bf16* __restrict__ vt,
                                                   bf16* __restrict__ o) {
  __shared__ __align__(16) bf16 k_s[2][64 * 128];  // [buf][key][feat]
  __shared__ __align__(16) bf16 vt_s[2][64 * 64];  // [buf][dv][key]
  __shared__ __align__(16) bf16 p_s[8 * 16 * 64];  // [wave][row][key]
  const int tid = threadIdx.x, wave = tid >> 5, lane = tid & 31;
  const int lh = lane & 15, ls = lane >> 4;
  const int qb = blockIdx.x & 7;
  const int bh = blockIdx.x >> 3;
  const int b = bh >> 4, h = bh & 15;
  const bf16* Q  = qp  + (size_t)bh * NN * 128;
  const bf16* Kp = kpg + (size_t)bh * NN * 128;
  const bf16* VT = vt  + (size_t)bh * 64 * NN;     // [dv][n]
  const int q0 = qb * 128 + wave * 16;

  v16bf qa[4];
  for (int c = 0; c < 4; c++) {
    const bf16* p = Q + (size_t)(q0 + lh) * 128 + c * 32 + ls * 8;
    FragU f; f.u[0] = *(const u32x4*)p; f.u[1] = *(const u32x4*)(p + 16);
    qa[c] = f.v;
  }

  float mstat[8], lstat[8];
  v8f oacc[4];
  for (int r = 0; r < 8; r++) { mstat[r] = -3.0e38f; lstat[r] = 0.f; }
  for (int t = 0; t < 4; t++)
    for (int r = 0; r < 8; r++) oacc[t][r] = 0.f;

  const int kr = tid >> 2;             // staging row 0..63
  const int kc = (tid & 3) * 32;       // K block col (halves)
  const int vc = (tid & 3) * 16;       // V block col (halves)

  auto stage_kv = [&](int kb, int bufi) {
    const bf16* srcK = Kp + (size_t)(kb + kr) * 128 + kc;
    bf16* dk = k_s[bufi] + kr * 128 + kc;
    async_b128(dk,      srcK);
    async_b128(dk + 8,  srcK + 8);
    async_b128(dk + 16, srcK + 16);
    async_b128(dk + 24, srcK + 24);
    const bf16* srcV = VT + (size_t)kr * NN + kb + vc;
    bf16* dv = vt_s[bufi] + kr * 64 + vc;
    async_b128(dv,     srcV);
    async_b128(dv + 8, srcV + 8);
  };

  stage_kv(0, 0);
  for (int kbi = 0; kbi < NN / 64; ++kbi) {
    const int bufi = kbi & 1;
    wait_async0();        // my block-kbi copies landed
    __syncthreads();      // all waves' copies landed; prior reads of buf^1 done
    if (kbi + 1 < NN / 64) stage_kv((kbi + 1) * 64, bufi ^ 1);

    // S = Q' K'^T  (16 q-rows x 64 keys)
    v8f s[4];
    for (int t = 0; t < 4; t++)
      for (int r = 0; r < 8; r++) s[t][r] = 0.f;
    for (int c = 0; c < 4; c++) {
      for (int t = 0; t < 4; t++) {
        const bf16* p = k_s[bufi] + (t * 16 + lh) * 128 + c * 32 + ls * 8;
        FragU f; f.u[0] = *(const u32x4*)p; f.u[1] = *(const u32x4*)(p + 16);
        s[t] = wmma_bf16(qa[c], f.v, s[t]);
      }
    }

    // online softmax update, write P to this wave's LDS strip
    bf16* pw = p_s + wave * (16 * 64);
    for (int r = 0; r < 8; r++) {
      float mx = fmaxf(fmaxf(s[0][r], s[1][r]), fmaxf(s[2][r], s[3][r]));
      for (int off = 8; off > 0; off >>= 1) mx = fmaxf(mx, __shfl_xor(mx, off, 32));
      const float newm = fmaxf(mstat[r], mx);
      const float corr = __expf(mstat[r] - newm);
      mstat[r] = newm;
      float rs = 0.f;
      for (int t = 0; t < 4; t++) {
        const float e = __expf(s[t][r] - newm);
        rs += e;
        pw[(ls * 8 + r) * 64 + t * 16 + lh] = (bf16)e;
      }
      for (int off = 8; off > 0; off >>= 1) rs += __shfl_xor(rs, off, 32);
      lstat[r] = lstat[r] * corr + rs;
      for (int t = 0; t < 4; t++) oacc[t][r] *= corr;
    }

    // O += P @ V  (K dim = 64 keys -> 2 chunks)
    for (int c = 0; c < 2; c++) {
      const bf16* pp = pw + lh * 64 + c * 32 + ls * 8;
      FragU fa; fa.u[0] = *(const u32x4*)pp; fa.u[1] = *(const u32x4*)(pp + 16);
      for (int t2 = 0; t2 < 4; t2++) {
        const bf16* pv = vt_s[bufi] + (t2 * 16 + lh) * 64 + c * 32 + ls * 8;
        FragU fb; fb.u[0] = *(const u32x4*)pv; fb.u[1] = *(const u32x4*)(pv + 16);
        oacc[t2] = wmma_bf16(fa.v, fb.v, oacc[t2]);
      }
    }
  }

  for (int r = 0; r < 8; r++) {
    const float inv = 1.0f / lstat[r];
    const int n = q0 + ls * 8 + r;
    for (int t2 = 0; t2 < 4; t2++) {
      const int d = h * 64 + t2 * 16 + lh;
      o[(size_t)(b * NN + n) * DD + d] = (bf16)(oacc[t2][r] * inv);
    }
  }
}

// ---------------------------------------------------------------------------
// SwiGLU: m[row, j] = silu(u[row, j]) * u[row, 4096 + j]
// ---------------------------------------------------------------------------
__global__ void swiglu_kernel(const bf16* __restrict__ u, bf16* __restrict__ m, int total) {
  int i = blockIdx.x * blockDim.x + threadIdx.x;
  const int stride = gridDim.x * blockDim.x;
  for (; i < total; i += stride) {
    const int row = i >> 12;
    const int j = i & 4095;
    const float a = (float)u[(size_t)row * 8192 + j];
    const float g = (float)u[(size_t)row * 8192 + 4096 + j];
    const float s = a / (1.f + __expf(-a));
    m[i] = (bf16)(s * g);
  }
}

// ---------------------------------------------------------------------------
// Host launcher
// ---------------------------------------------------------------------------
extern "C" void kernel_launch(void* const* d_in, const int* in_sizes, int n_in,
                              void* d_out, int out_size, void* d_ws, size_t ws_size,
                              hipStream_t stream) {
  (void)in_sizes; (void)n_in; (void)out_size; (void)ws_size;
  const float* x        = (const float*)d_in[0];
  const float* coords   = (const float*)d_in[1];
  const int*   rope_pos = (const int*)d_in[2];
  const float* ln1_w    = (const float*)d_in[3];
  const float* ln1_b    = (const float*)d_in[4];
  const float* qkv_w    = (const float*)d_in[5];
  const float* qkv_b    = (const float*)d_in[6];
  const float* q_norm_w = (const float*)d_in[7];
  const float* k_norm_w = (const float*)d_in[8];
  const float* sq_w     = (const float*)d_in[9];
  const float* sk_w     = (const float*)d_in[10];
  const float* out_w    = (const float*)d_in[11];
  const float* out_b    = (const float*)d_in[12];
  const float* ls1_g    = (const float*)d_in[13];
  const float* ln2_w    = (const float*)d_in[14];
  const float* ln2_b    = (const float*)d_in[15];
  const float* w12_w    = (const float*)d_in[16];
  const float* w12_b    = (const float*)d_in[17];
  const float* wo_w     = (const float*)d_in[18];
  const float* wo_b     = (const float*)d_in[19];
  const float* ls2_g    = (const float*)d_in[20];

  char* ws = (char*)d_ws;
  // Offsets (bytes); u_bf reuses dead qkv_f+qp, m_bf reuses dead kp+vt+o.
  bf16*  wq_bf   = (bf16*)(ws + 0);          //  6,291,456  (3072x1024)
  bf16*  wout_bf = (bf16*)(ws + 6291456);    //  2,097,152  (1024x1024)
  bf16*  w12_bf  = (bf16*)(ws + 8388608);    // 16,777,216  (8192x1024)
  bf16*  wo_bf   = (bf16*)(ws + 25165824);   //  8,388,608  (1024x4096)
  bf16*  h_bf    = (bf16*)(ws + 33554432);   //  8,388,608  (4096x1024)
  float* qkv_f   = (float*)(ws + 41943040);  // 50,331,648  (4096x3072)
  bf16*  qp      = (bf16*)(ws + 92274688);   // 16,777,216  (64x1024x128)
  bf16*  kp      = (bf16*)(ws + 109051904);  // 16,777,216
  bf16*  vt_bf   = (bf16*)(ws + 125829120);  //  8,388,608  (64x64x1024, V^T)
  bf16*  o_bf    = (bf16*)(ws + 134217728);  //  8,388,608  (4096x1024)
  float* x1      = (float*)(ws + 142606336); // 16,777,216  (4096x1024)
  bf16*  u_bf    = (bf16*)(ws + 41943040);   // 67,108,864  (overlaps qkv_f+qp)
  bf16*  m_bf    = (bf16*)(ws + 109051904);  // 33,554,432  (overlaps kp+vt+o)

  // 1) weights -> bf16
  f2bf_kernel<<<4096, 256, 0, stream>>>(qkv_w, wq_bf, 3072 * 1024);
  f2bf_kernel<<<2048, 256, 0, stream>>>(out_w, wout_bf, 1024 * 1024);
  f2bf_kernel<<<8192, 256, 0, stream>>>(w12_w, w12_bf, 8192 * 1024);
  f2bf_kernel<<<4096, 256, 0, stream>>>(wo_w, wo_bf, 1024 * 4096);

  // 2) LN1
  ln_kernel<<<BNROWS, 256, 0, stream>>>(x, ln1_w, ln1_b, h_bf);

  // 3) QKV GEMM: [4096,1024] x [3072,1024]^T -> f32
  gemm_bf16_kernel<0><<<dim3(24, 32), 256, 0, stream>>>(
      h_bf, wq_bf, qkv_b, qkv_f, nullptr, nullptr, BNROWS, 3072, 1024);

  // 4) RMSNorm + RoPE + spatial-bias feature concat (+ V transpose)
  qkv_post_kernel<<<8192, 256, 0, stream>>>(qkv_f, rope_pos, q_norm_w, k_norm_w,
                                            coords, sq_w, sk_w, qp, kp, vt_bf);

  // 5) flash attention
  attn_kernel<<<512, 256, 0, stream>>>(qp, kp, vt_bf, o_bf);

  // 6) out-proj + residual layerscale: x1 = x + ls1 * (o @ out_w^T + out_b)
  gemm_bf16_kernel<2><<<dim3(8, 32), 256, 0, stream>>>(
      o_bf, wout_bf, out_b, x1, x, ls1_g, BNROWS, 1024, 1024);

  // 7) LN2
  ln_kernel<<<BNROWS, 256, 0, stream>>>(x1, ln2_w, ln2_b, h_bf);

  // 8) MLP up: u = h2 @ w12^T + b  (bf16 out)
  gemm_bf16_kernel<1><<<dim3(64, 32), 256, 0, stream>>>(
      h_bf, w12_bf, w12_b, u_bf, nullptr, nullptr, BNROWS, 8192, 1024);

  // 9) SwiGLU
  swiglu_kernel<<<8192, 256, 0, stream>>>(u_bf, m_bf, BNROWS * 4096);

  // 10) MLP down + residual layerscale -> d_out (f32)
  gemm_bf16_kernel<2><<<dim3(8, 32), 256, 0, stream>>>(
      m_bf, wo_bf, wo_b, (float*)d_out, x1, ls2_g, BNROWS, 1024, 4096);
}
